// InfoNCELoss_69320772157801
// MI455X (gfx1250) — matrix operand order repaired and checked by reading, not compiled, not run on hardware
//
#include <hip/hip_runtime.h>
#include <cstdint>
#include <cstddef>

#define B_N 4096
#define D_K 2048
#define MT 128
#define NT 128
#define KT 32
#define LDT 40   // padded LDS row stride in halves (80B) -> conflict-free frag reads

typedef __attribute__((ext_vector_type(16))) _Float16 v16h;
typedef __attribute__((ext_vector_type(8)))  _Float16 v8h;
typedef __attribute__((ext_vector_type(8)))  float    v8f;
typedef __attribute__((ext_vector_type(4)))  float    v4f;

// ---------------- 1) row-wise L2 normalize -> f16 ----------------
__global__ __launch_bounds__(256) void rownorm_kernel(const float* __restrict__ cond,
                                                      const float* __restrict__ sol,
                                                      _Float16* __restrict__ ch,
                                                      _Float16* __restrict__ sh) {
    const int bid = blockIdx.x;
    const float*   src = (bid < B_N) ? cond : sol;
    _Float16*      dst = (bid < B_N) ? ch   : sh;
    const int row = bid & (B_N - 1);
    const int t = threadIdx.x;
    const float* rp = src + (size_t)row * D_K + t * 8;
    _Float16*    wp = dst + (size_t)row * D_K + t * 8;

    v4f x0 = *(const v4f*)(rp);
    v4f x1 = *(const v4f*)(rp + 4);
    float ss = 0.f;
#pragma unroll
    for (int i = 0; i < 4; ++i) ss += x0[i]*x0[i] + x1[i]*x1[i];
#pragma unroll
    for (int off = 16; off > 0; off >>= 1) ss += __shfl_xor(ss, off, 32);

    __shared__ float wsum[8];
    __shared__ float tot_s;
    const int lane = t & 31, wid = t >> 5;
    if (lane == 0) wsum[wid] = ss;
    __syncthreads();
    if (t == 0) {
        float s = 0.f;
#pragma unroll
        for (int i = 0; i < 8; ++i) s += wsum[i];
        tot_s = s;
    }
    __syncthreads();
    const float inv = 1.f / fmaxf(sqrtf(tot_s), 1e-12f);

    v8h h;
#pragma unroll
    for (int i = 0; i < 4; ++i) {
        h[i]     = (_Float16)(x0[i] * inv);
        h[4 + i] = (_Float16)(x1[i] * inv);
    }
    *(v8h*)wp = h;
}

// ---------------- 2) zero accumulators (every call: graph replay) ----------------
__global__ void zero_kernel(float* __restrict__ p, int n) {
    int i = blockIdx.x * blockDim.x + threadIdx.x;
    if (i < n) p[i] = 0.f;
}

// ---------------- 3) diagonal: relu(c_i . s_i) * exp(T) ----------------
__global__ __launch_bounds__(256) void diag_kernel(const _Float16* __restrict__ ch,
                                                   const _Float16* __restrict__ sh,
                                                   const float* __restrict__ temp,
                                                   float* __restrict__ diag) {
    const int row = blockIdx.x;
    const int t = threadIdx.x;
    v8h a = *(const v8h*)(ch + (size_t)row * D_K + t * 8);
    v8h b = *(const v8h*)(sh + (size_t)row * D_K + t * 8);
    float s = 0.f;
#pragma unroll
    for (int i = 0; i < 8; ++i) s += (float)a[i] * (float)b[i];
#pragma unroll
    for (int off = 16; off > 0; off >>= 1) s += __shfl_xor(s, off, 32);

    __shared__ float wsum[8];
    const int lane = t & 31, wid = t >> 5;
    if (lane == 0) wsum[wid] = s;
    __syncthreads();
    if (t == 0) {
        float tot = 0.f;
#pragma unroll
        for (int i = 0; i < 8; ++i) tot += wsum[i];
        diag[row] = fmaxf(tot, 0.f) * expf(temp[0]);
    }
}

// ---------------- 4) fused WMMA GEMM + row/col exp-sum (double-buffered) ----------------
__global__ __launch_bounds__(256) void gemm_lse_kernel(const _Float16* __restrict__ A,
                                                       const _Float16* __restrict__ Bm,
                                                       const float* __restrict__ temp,
                                                       float* __restrict__ rsum,
                                                       float* __restrict__ csum) {
    __shared__ _Float16 As[2][MT * LDT];
    __shared__ _Float16 Bs[2][NT * LDT];
    __shared__ float rs[MT];
    __shared__ float cs[NT];

    const int t    = threadIdx.x;
    const int m0   = blockIdx.y * MT;
    const int n0   = blockIdx.x * NT;
    const int lane = t & 31;
    const int w    = t >> 5;      // wave 0..7
    const int wm   = w & 3;       // 32-row group
    const int wn   = w >> 2;      // 64-col group
    const int half = lane >> 4;   // 0: lanes 0-15, 1: lanes 16-31
    const int ln   = lane & 15;

    v8f acc[2][4];
#pragma unroll
    for (int mi = 0; mi < 2; ++mi)
#pragma unroll
        for (int ni = 0; ni < 4; ++ni)
            acc[mi][ni] = (v8f){0.f,0.f,0.f,0.f,0.f,0.f,0.f,0.f};

    // tile loaders: thread t -> row t/2, 16-half (t&1)
    const int rr = t >> 1;
    const int hh = (t & 1) * 16;
    const _Float16* gA = A  + (size_t)(m0 + rr) * D_K + hh;
    const _Float16* gB = Bm + (size_t)(n0 + rr) * D_K + hh;
    const int stoff = rr * LDT + hh;

    // 8 WMMAs from one LDS buffer pair
    auto compute = [&](const _Float16* Asb, const _Float16* Bsb) {
        v16h af[2], bf[4];
#pragma unroll
        for (int mi = 0; mi < 2; ++mi) {
            const _Float16* ap = &Asb[(wm * 32 + mi * 16 + ln) * LDT];
            v8h lo = *(const v8h*)(ap + half * 8);
            v8h hi = *(const v8h*)(ap + 16 + half * 8);
            v16h f;
#pragma unroll
            for (int i = 0; i < 8; ++i) { f[i] = lo[i]; f[8 + i] = hi[i]; }
            af[mi] = f;
        }
#pragma unroll
        for (int ni = 0; ni < 4; ++ni) {
            const _Float16* bp = &Bsb[(wn * 64 + ni * 16 + ln) * LDT];
            v8h lo = *(const v8h*)(bp + half * 8);
            v8h hi = *(const v8h*)(bp + 16 + half * 8);
            v16h f;
#pragma unroll
            for (int i = 0; i < 8; ++i) { f[i] = lo[i]; f[8 + i] = hi[i]; }
            bf[ni] = f;
        }
#pragma unroll
        for (int mi = 0; mi < 2; ++mi)
#pragma unroll
            for (int ni = 0; ni < 4; ++ni)
                acc[mi][ni] = __builtin_amdgcn_wmma_f32_16x16x32_f16(
                    false, af[mi], false, bf[ni], (short)0, acc[mi][ni], false, false);
    };

    // ---- prologue: stage tile k=0 into buffer 0 ----
    {
        v8h a0 = *(const v8h*)(gA);
        v8h a1 = *(const v8h*)(gA + 8);
        v8h b0 = *(const v8h*)(gB);
        v8h b1 = *(const v8h*)(gB + 8);
        *(v8h*)(&As[0][stoff])     = a0;
        *(v8h*)(&As[0][stoff + 8]) = a1;
        *(v8h*)(&Bs[0][stoff])     = b0;
        *(v8h*)(&Bs[0][stoff + 8]) = b1;
    }
    __syncthreads();

    // ---- main loop, 2x unrolled ping-pong: one barrier per K-step ----
    for (int kk = 0; kk < D_K; kk += 2 * KT) {
        if (kk + 4 * KT < D_K) {           // long-range gfx1250 global_prefetch_b8
            __builtin_prefetch(gA + kk + 4 * KT, 0, 1);
            __builtin_prefetch(gB + kk + 4 * KT, 0, 1);
        }
        // phase A: loads for tile kk+KT in flight while computing buffer 0
        {
            v8h a0 = *(const v8h*)(gA + kk + KT);
            v8h a1 = *(const v8h*)(gA + kk + KT + 8);
            v8h b0 = *(const v8h*)(gB + kk + KT);
            v8h b1 = *(const v8h*)(gB + kk + KT + 8);
            compute(As[0], Bs[0]);
            *(v8h*)(&As[1][stoff])     = a0;
            *(v8h*)(&As[1][stoff + 8]) = a1;
            *(v8h*)(&Bs[1][stoff])     = b0;
            *(v8h*)(&Bs[1][stoff + 8]) = b1;
        }
        __syncthreads();
        // phase B: loads for tile kk+2*KT in flight while computing buffer 1
        {
            const bool more = (kk + 2 * KT) < D_K;
            v8h a0, a1, b0, b1;
            if (more) {
                a0 = *(const v8h*)(gA + kk + 2 * KT);
                a1 = *(const v8h*)(gA + kk + 2 * KT + 8);
                b0 = *(const v8h*)(gB + kk + 2 * KT);
                b1 = *(const v8h*)(gB + kk + 2 * KT + 8);
            }
            compute(As[1], Bs[1]);
            if (more) {
                *(v8h*)(&As[0][stoff])     = a0;
                *(v8h*)(&As[0][stoff + 8]) = a1;
                *(v8h*)(&Bs[0][stoff])     = b0;
                *(v8h*)(&Bs[0][stoff + 8]) = b1;
            }
        }
        __syncthreads();
    }

    // ---- epilogue: e = exp(relu(x)*s - s); accumulate row & col sums ----
    const float scale = expf(temp[0]);
    if (t < MT) { rs[t] = 0.f; cs[t] = 0.f; }
    __syncthreads();

#pragma unroll
    for (int mi = 0; mi < 2; ++mi) {
#pragma unroll
        for (int ni = 0; ni < 4; ++ni) {
            float ev[8];
            float colv = 0.f;
#pragma unroll
            for (int r8 = 0; r8 < 8; ++r8) {
                float x = fmaxf(acc[mi][ni][r8], 0.f) * scale;
                float e = expf(x - scale);
                ev[r8] = e;
                colv += e;   // fixed column N=ln, rows r8+8*half
            }
            colv += __shfl_xor(colv, 16, 32);            // fold both 8-row halves
            if (half == 0) atomicAdd(&cs[wn * 64 + ni * 16 + ln], colv);
#pragma unroll
            for (int r8 = 0; r8 < 8; ++r8) {
                float e = ev[r8];                        // butterfly over N (lanes 0-15 of each half)
                e += __shfl_xor(e, 1, 32);
                e += __shfl_xor(e, 2, 32);
                e += __shfl_xor(e, 4, 32);
                e += __shfl_xor(e, 8, 32);
                if (ln == 0)
                    atomicAdd(&rs[wm * 32 + mi * 16 + r8 + half * 8], e);
            }
        }
    }
    __syncthreads();
    if (t < MT) atomicAdd(&rsum[m0 + t], rs[t]);
    else        atomicAdd(&csum[n0 + (t - MT)], cs[t - MT]);
}

// ---------------- 5) final scalar reduction ----------------
__global__ __launch_bounds__(256) void final_kernel(const float* __restrict__ rsum,
                                                    const float* __restrict__ csum,
                                                    const float* __restrict__ diag,
                                                    const float* __restrict__ temp,
                                                    float* __restrict__ out) {
    const float scale = expf(temp[0]);
    float acc = 0.f;
    for (int i = threadIdx.x; i < B_N; i += 256) {
        // LSE_row = scale + log(sum exp(x - scale)); same for columns
        acc += (scale + logf(rsum[i])) + (scale + logf(csum[i])) - 2.f * diag[i];
    }
#pragma unroll
    for (int off = 16; off > 0; off >>= 1) acc += __shfl_xor(acc, off, 32);
    __shared__ float wsum[8];
    const int lane = threadIdx.x & 31, wid = threadIdx.x >> 5;
    if (lane == 0) wsum[wid] = acc;
    __syncthreads();
    if (threadIdx.x == 0) {
        float s = 0.f;
#pragma unroll
        for (int i = 0; i < 8; ++i) s += wsum[i];
        out[0] = s * 0.5f / (float)B_N;
    }
}

extern "C" void kernel_launch(void* const* d_in, const int* in_sizes, int n_in,
                              void* d_out, int out_size, void* d_ws, size_t ws_size,
                              hipStream_t stream) {
    (void)in_sizes; (void)n_in; (void)out_size; (void)ws_size;
    const float* cond = (const float*)d_in[0];
    const float* sol  = (const float*)d_in[1];
    const float* temp = (const float*)d_in[2];
    float* out = (float*)d_out;

    _Float16* ch   = (_Float16*)d_ws;                  // 16 MB
    _Float16* sh   = ch + (size_t)B_N * D_K;           // 16 MB
    float*    rsum = (float*)(sh + (size_t)B_N * D_K); // 16 KB
    float*    csum = rsum + B_N;                       // 16 KB
    float*    diag = csum + B_N;                       // 16 KB

    rownorm_kernel<<<2 * B_N, 256, 0, stream>>>(cond, sol, ch, sh);
    zero_kernel<<<(2 * B_N + 255) / 256, 256, 0, stream>>>(rsum, 2 * B_N);
    diag_kernel<<<B_N, 256, 0, stream>>>(ch, sh, temp, diag);
    dim3 grid(B_N / NT, B_N / MT);
    gemm_lse_kernel<<<grid, 256, 0, stream>>>(ch, sh, temp, rsum, csum);
    final_kernel<<<1, 256, 0, stream>>>(rsum, csum, diag, temp, out);
}
